// MultiHeadedAttention_33311766348367
// MI455X (gfx1250) — matrix operand order repaired and checked by reading, not compiled, not run on hardware
//
#include <hip/hip_runtime.h>

// ---------------------------------------------------------------------------
// MI455X (gfx1250) multi-head attention forward.
// All GEMMs on v_wmma_f32_16x16x32_bf16 (fp32 accumulate).
// GEMM: 128x128 block tile, K-step 64, double-buffered LDS (1 barrier/stage).
// Attention: flash-style; K tile staged via async global->LDS (ASYNCcnt),
// V^T staged via VALU transpose, Q fragments pinned in registers.
// ---------------------------------------------------------------------------

#define D_MODEL  1024
#define N_HEADS  16
#define HEAD_DIM 64
#define BATCH    4
#define SEQ      2048
#define M_TOTAL  (BATCH * SEQ)   // 8192

typedef __attribute__((ext_vector_type(16))) __bf16 v16bf;
typedef __attribute__((ext_vector_type(8)))  __bf16 v8bf;
typedef __attribute__((ext_vector_type(4)))  __bf16 v4bf;
typedef __attribute__((ext_vector_type(8)))  float  v8f;

union Frag16 { v16bf v; v8bf h[2]; };

__device__ __forceinline__ v8f wmma_bf16(v16bf a, v16bf b, v8f c) {
  // (neg_a, A, neg_b, B, c_mod, C, reuse_a, reuse_b)
  return __builtin_amdgcn_wmma_f32_16x16x32_bf16(false, a, false, b, (short)0, c,
                                                 false, false);
}

__device__ __forceinline__ v8f vzero8() {
  v8f z;
#pragma unroll
  for (int i = 0; i < 8; ++i) z[i] = 0.0f;
  return z;
}

// A-matrix fragment (16x32 slice, 16-bit): lane L<16 holds row L, K chunks
// {0..7, 16..23}; lane L+16 holds row L, K chunks {8..15, 24..31}.
__device__ __forceinline__ v16bf load_fragA(const __bf16* base, int row0,
                                            int stride, int k0, int lane) {
  const int r  = row0 + (lane & 15);
  const int kk = k0 + ((lane & 16) >> 1);   // +8 for upper half-wave
  Frag16 f;
  f.h[0] = *(const v8bf*)(base + (size_t)r * stride + kk);
  f.h[1] = *(const v8bf*)(base + (size_t)r * stride + kk + 16);
  return f.v;
}

// B-matrix fragment (32x16, 16-bit), stored [n][k] with `stride`:
// lane L<16 holds column L, K {0..7, 16..23}; lane L+16 holds column L, K {8..15, 24..31}.
__device__ __forceinline__ v16bf load_fragB(const __bf16* base, int col0,
                                            int stride, int k0, int lane) {
  const int n  = col0 + (lane & 15);
  const int kk = k0 + ((lane & 16) >> 1);   // +8 for upper half-wave
  Frag16 f;
  f.h[0] = *(const v8bf*)(base + (size_t)n * stride + kk);
  f.h[1] = *(const v8bf*)(base + (size_t)n * stride + kk + 16);
  return f.v;
}

// CDNA5 async global->LDS copy of one 16-byte chunk (tracked by ASYNCcnt).
// LDS destination address = low 32 bits of the shared-memory flat address
// (ISA: LDS aperture addresses truncate to addr[31:0]).
__device__ __forceinline__ void async_copy_b128(const void* gsrc, void* lds_dst) {
  const unsigned lds_off = (unsigned)(uintptr_t)lds_dst;
  asm volatile("global_load_async_to_lds_b128 %0, %1, off"
               :
               : "v"(lds_off), "v"(gsrc)
               : "memory");
}

__device__ __forceinline__ void wait_asynccnt0() {
  asm volatile("s_wait_asynccnt 0" ::: "memory");
}

// ---------------------------------------------------------------------------
// C[M,N] = A[M,K] @ W[K,N] + bias.  A f32 or bf16 (cast on the fly), W/bias
// f32, out bf16 or f32.  128x128 tile, K-step 64, double-buffered LDS.
// ---------------------------------------------------------------------------
template <bool A_F32, bool OUT_F32>
__global__ __launch_bounds__(256) void gemm_bias_kernel(
    const void* __restrict__ Av, const float* __restrict__ W,
    const float* __restrict__ bias, void* __restrict__ Ov, int M, int N, int K) {
  constexpr int LS = 72;                 // padded LDS stride (elements)
  __shared__ __bf16 As[2 * 128 * LS];    // [buf][row][k]
  __shared__ __bf16 Bs[2 * 128 * LS];    // [buf][n][k]  (W transposed)

  const int t    = threadIdx.x;
  const int lane = t & 31;
  const int wave = t >> 5;
  const int wm   = (wave & 3) * 32;
  const int wn   = (wave >> 2) * 64;
  const int m0   = blockIdx.y * 128;
  const int n0   = blockIdx.x * 128;
  const int NS   = K / 64;

  v8f acc[2][4];
#pragma unroll
  for (int mt = 0; mt < 2; ++mt)
#pragma unroll
    for (int nt = 0; nt < 4; ++nt) acc[mt][nt] = vzero8();

  auto load_stage = [&](int s, int buf) {
    const int k0 = s * 64;
    __bf16* Ab = &As[buf * 128 * LS];
    __bf16* Bb = &Bs[buf * 128 * LS];
    // ---- A tile: 128x64 -> bf16 ----
    if (A_F32) {
      const float* A = (const float*)Av;
#pragma unroll
      for (int i = 0; i < 8; ++i) {
        const int idx = t + i * 256;      // 0..2047 float4 chunks
        const int row = idx >> 4;         // 16 chunks per row
        const int c4  = (idx & 15) << 2;
        const float4 sv = *(const float4*)(A + (size_t)(m0 + row) * K + k0 + c4);
        __bf16* d = &Ab[row * LS + c4];
        d[0] = (__bf16)sv.x; d[1] = (__bf16)sv.y;
        d[2] = (__bf16)sv.z; d[3] = (__bf16)sv.w;
      }
      if (s + 2 < NS)   // pull stage s+2 toward cache while matrix pipe is busy
        __builtin_prefetch(A + (size_t)(m0 + (t >> 1)) * K + (s + 2) * 64, 0, 0);
    } else {
      const __bf16* A = (const __bf16*)Av;
#pragma unroll
      for (int i = 0; i < 8; ++i) {
        const int idx = t + i * 256;
        const int row = idx >> 4;
        const int c4  = (idx & 15) << 2;
        const v4bf sv = *(const v4bf*)(A + (size_t)(m0 + row) * K + k0 + c4);
        *(v4bf*)&Ab[row * LS + c4] = sv;
      }
      if (s + 2 < NS)
        __builtin_prefetch(A + (size_t)(m0 + (t >> 1)) * K + (s + 2) * 64, 0, 0);
    }
    // ---- W tile: 64x128, transpose into Bb[n][k] ----
#pragma unroll
    for (int i = 0; i < 8; ++i) {
      const int idx = t + i * 256;        // 0..2047 float4 chunks
      const int kr  = idx >> 5;           // 0..63
      const int n4  = (idx & 31) << 2;
      const float4 sv = *(const float4*)(W + (size_t)(k0 + kr) * N + n0 + n4);
      Bb[(n4 + 0) * LS + kr] = (__bf16)sv.x;
      Bb[(n4 + 1) * LS + kr] = (__bf16)sv.y;
      Bb[(n4 + 2) * LS + kr] = (__bf16)sv.z;
      Bb[(n4 + 3) * LS + kr] = (__bf16)sv.w;
    }
    if (s + 2 < NS)
      __builtin_prefetch(W + (size_t)((s + 2) * 64 + (t >> 2)) * N + n0, 0, 0);
  };

  auto compute_stage = [&](int buf) {
    const __bf16* Ab = &As[buf * 128 * LS];
    const __bf16* Bb = &Bs[buf * 128 * LS];
#pragma unroll
    for (int kc = 0; kc < 2; ++kc) {
      v16bf a[2], b[4];
#pragma unroll
      for (int mt = 0; mt < 2; ++mt)
        a[mt] = load_fragA(Ab, wm + mt * 16, LS, kc * 32, lane);
#pragma unroll
      for (int nt = 0; nt < 4; ++nt)
        b[nt] = load_fragB(Bb, wn + nt * 16, LS, kc * 32, lane);
#pragma unroll
      for (int mt = 0; mt < 2; ++mt)
#pragma unroll
        for (int nt = 0; nt < 4; ++nt)
          acc[mt][nt] = wmma_bf16(a[mt], b[nt], acc[mt][nt]);
    }
  };

  load_stage(0, 0);
  __syncthreads();
  for (int s = 0; s < NS; ++s) {
    if (s + 1 < NS) load_stage(s + 1, (s + 1) & 1);  // fill other buffer
    compute_stage(s & 1);                            // consume current buffer
    __syncthreads();
  }

  // ---- epilogue: bias + store ----
  const int hi8 = (lane & 16) >> 1;
  const int nl  = lane & 15;
#pragma unroll
  for (int mt = 0; mt < 2; ++mt)
#pragma unroll
    for (int nt = 0; nt < 4; ++nt) {
      const int n    = n0 + wn + nt * 16 + nl;
      const float bb = bias[n];
#pragma unroll
      for (int r = 0; r < 8; ++r) {
        const int m   = m0 + wm + mt * 16 + r + hi8;
        const float v = acc[mt][nt][r] + bb;
        if (OUT_F32)
          ((float*)Ov)[(size_t)m * N + n] = v;
        else
          ((__bf16*)Ov)[(size_t)m * N + n] = (__bf16)v;
      }
    }
}

// ---------------------------------------------------------------------------
// Flash attention over one (b, h, 128-row Q tile).  8 waves x 16 Q rows.
// K tile staged via async global->LDS; V^T via VALU transpose; Q in regs.
// ---------------------------------------------------------------------------
__global__ __launch_bounds__(256) void attn_kernel(
    const void* __restrict__ Qv, const void* __restrict__ Kv,
    const void* __restrict__ Vv, void* __restrict__ Cv) {
  __shared__ __bf16 Ks[64 * 72];        // K tile: [key][d]        (B-frag ready)
  __shared__ __bf16 Vt[64 * 72];        // V tile transposed: [d][key]
  __shared__ __bf16 Ps[8 * 16 * 72];    // per-wave P staging: [row][key]

  const int t    = threadIdx.x;
  const int lane = t & 31;
  const int wave = t >> 5;
  const int qt   = blockIdx.x;          // 0..15
  const int h    = blockIdx.y;          // 0..15
  const int b    = blockIdx.z;          // 0..3

  const size_t bhoff = ((size_t)b * SEQ) * D_MODEL + (size_t)h * HEAD_DIM;
  const __bf16* Qb = (const __bf16*)Qv + bhoff;
  const __bf16* Kb = (const __bf16*)Kv + bhoff;
  const __bf16* Vb = (const __bf16*)Vv + bhoff;
  __bf16*       Cb = (__bf16*)Cv + bhoff;

  const int q0 = qt * 128 + wave * 16;
  __bf16* Pw = &Ps[wave * 16 * 72];

  // Q fragments (A layout) kept in registers for the whole kernel.
  v16bf qf[2];
#pragma unroll
  for (int ds = 0; ds < 2; ++ds)
    qf[ds] = load_fragA(Qb, q0, D_MODEL, ds * 32, lane);

  v8f o[4];
#pragma unroll
  for (int dt = 0; dt < 4; ++dt) o[dt] = vzero8();
  float rowmax[8], rowsum[8];
#pragma unroll
  for (int r = 0; r < 8; ++r) { rowmax[r] = -1e30f; rowsum[r] = 0.0f; }

  const float scale = 0.125f;           // 1/sqrt(64)
  const int hi8 = (lane & 16) >> 1;
  const int nl  = lane & 15;

  for (int j = 0; j < SEQ / 64; ++j) {
    if (j + 1 < SEQ / 64) {             // prefetch next K/V tile rows
      const size_t nrow = (size_t)((j + 1) * 64 + (t >> 2)) * D_MODEL;
      __builtin_prefetch(Kb + nrow, 0, 0);
      __builtin_prefetch(Vb + nrow, 0, 0);
    }
    __syncthreads();                    // protect Ks/Vt from previous iteration
    // ---- stage K tile [key][d] (async DMA) and V^T tile [d][key] (VALU) ----
#pragma unroll
    for (int i = 0; i < 2; ++i) {
      const int idx = t + i * 256;      // 0..511 8-elem chunks
      const int key = idx >> 3;         // 0..63
      const int d8  = (idx & 7) << 3;   // 0,8,..,56
      const size_t grow = (size_t)(j * 64 + key) * D_MODEL + d8;
      async_copy_b128(Kb + grow, &Ks[key * 72 + d8]);   // no VGPR bounce
      const v8bf vv = *(const v8bf*)(Vb + grow);
#pragma unroll
      for (int e = 0; e < 8; ++e) Vt[(d8 + e) * 72 + key] = vv[e];
    }
    wait_asynccnt0();                   // K tile landed in LDS
    __syncthreads();

    // ---- scores S = Q K^T (16 x 64 per wave) ----
    v8f s[4];
#pragma unroll
    for (int nt = 0; nt < 4; ++nt) s[nt] = vzero8();
#pragma unroll
    for (int nt = 0; nt < 4; ++nt)
#pragma unroll
      for (int ds = 0; ds < 2; ++ds) {
        const v16bf kf = load_fragB(Ks, nt * 16, 72, ds * 32, lane);
        s[nt] = wmma_bf16(qf[ds], kf, s[nt]);
      }

    // ---- online softmax (row stats per VGPR slot, 16-lane reductions) ----
    float mnew[8];
#pragma unroll
    for (int r = 0; r < 8; ++r) mnew[r] = rowmax[r];
#pragma unroll
    for (int nt = 0; nt < 4; ++nt) {
      s[nt] *= scale;
#pragma unroll
      for (int r = 0; r < 8; ++r) mnew[r] = fmaxf(mnew[r], s[nt][r]);
    }
#pragma unroll
    for (int mask = 8; mask >= 1; mask >>= 1)
#pragma unroll
      for (int r = 0; r < 8; ++r)
        mnew[r] = fmaxf(mnew[r], __shfl_xor(mnew[r], mask, 32));

    float corr[8], psum[8];
#pragma unroll
    for (int r = 0; r < 8; ++r) {
      corr[r]   = __expf(rowmax[r] - mnew[r]);
      rowmax[r] = mnew[r];
      psum[r]   = 0.0f;
    }
#pragma unroll
    for (int nt = 0; nt < 4; ++nt)
#pragma unroll
      for (int r = 0; r < 8; ++r) {
        const float p = __expf(s[nt][r] - rowmax[r]);
        psum[r] += p;
        Pw[(r + hi8) * 72 + nt * 16 + nl] = (__bf16)p;
      }
#pragma unroll
    for (int mask = 8; mask >= 1; mask >>= 1)
#pragma unroll
      for (int r = 0; r < 8; ++r) psum[r] += __shfl_xor(psum[r], mask, 32);
#pragma unroll
    for (int r = 0; r < 8; ++r) rowsum[r] = rowsum[r] * corr[r] + psum[r];
#pragma unroll
    for (int dt = 0; dt < 4; ++dt)
#pragma unroll
      for (int r = 0; r < 8; ++r) o[dt][r] *= corr[r];

    // P just written to per-wave LDS; drain DS counter before reading back.
    asm volatile("s_wait_dscnt 0" ::: "memory");

    // ---- O += P V  (16x64 += 16x64 * 64x64) ----
#pragma unroll
    for (int kk = 0; kk < 2; ++kk) {
      const v16bf pf = load_fragA(Pw, 0, 72, kk * 32, lane);
#pragma unroll
      for (int dt = 0; dt < 4; ++dt) {
        const v16bf vf = load_fragB(Vt, dt * 16, 72, kk * 32, lane);
        o[dt] = wmma_bf16(pf, vf, o[dt]);
      }
    }
  }

  // ---- normalize and store ctx (bf16, concat layout [B,S,D]) ----
#pragma unroll
  for (int r = 0; r < 8; ++r) rowsum[r] = 1.0f / rowsum[r];
#pragma unroll
  for (int dt = 0; dt < 4; ++dt)
#pragma unroll
    for (int r = 0; r < 8; ++r) {
      const int q = q0 + r + hi8;
      const int d = dt * 16 + nl;
      Cb[(size_t)q * D_MODEL + d] = (__bf16)(o[dt][r] * rowsum[r]);
    }
}

// ---------------------------------------------------------------------------
extern "C" void kernel_launch(void* const* d_in, const int* in_sizes, int n_in,
                              void* d_out, int out_size, void* d_ws,
                              size_t ws_size, hipStream_t stream) {
  const float* q  = (const float*)d_in[0];
  const float* k  = (const float*)d_in[1];
  const float* v  = (const float*)d_in[2];
  const float* Wq = (const float*)d_in[3];
  const float* bq = (const float*)d_in[4];
  const float* Wk = (const float*)d_in[5];
  const float* bk = (const float*)d_in[6];
  const float* Wv = (const float*)d_in[7];
  const float* bv = (const float*)d_in[8];
  const float* Wo = (const float*)d_in[9];
  const float* bo = (const float*)d_in[10];

  // Workspace: 4 bf16 [8192 x 1024] buffers = 64 MiB.
  const size_t elems = (size_t)M_TOTAL * D_MODEL;
  char* wsb = (char*)d_ws;
  void* Qp  = (void*)(wsb + 0 * elems * 2);
  void* Kp  = (void*)(wsb + 1 * elems * 2);
  void* Vp  = (void*)(wsb + 2 * elems * 2);
  void* Ctx = (void*)(wsb + 3 * elems * 2);

  const dim3 gg(D_MODEL / 128, M_TOTAL / 128);   // (8, 64)
  gemm_bias_kernel<true, false><<<gg, 256, 0, stream>>>(q, Wq, bq, Qp,
                                                        M_TOTAL, D_MODEL, D_MODEL);
  gemm_bias_kernel<true, false><<<gg, 256, 0, stream>>>(k, Wk, bk, Kp,
                                                        M_TOTAL, D_MODEL, D_MODEL);
  gemm_bias_kernel<true, false><<<gg, 256, 0, stream>>>(v, Wv, bv, Vp,
                                                        M_TOTAL, D_MODEL, D_MODEL);

  const dim3 ga(SEQ / 128, N_HEADS, BATCH);      // (16, 16, 4)
  attn_kernel<<<ga, 256, 0, stream>>>(Qp, Kp, Vp, Ctx);

  gemm_bias_kernel<false, true><<<gg, 256, 0, stream>>>(Ctx, Wo, bo, d_out,
                                                        M_TOTAL, D_MODEL, D_MODEL);
}
